// CSSNet_28484223107291
// MI455X (gfx1250) — matrix operand order repaired and checked by reading, not compiled, not run on hardware
//
#include <hip/hip_runtime.h>
#include <hip/hip_bf16.h>

typedef __attribute__((ext_vector_type(16))) _Float16 v16h;
typedef __attribute__((ext_vector_type(8)))  _Float16 v8h;
typedef __attribute__((ext_vector_type(8)))  float    v8f;
typedef long long i64;

#define HSI 31
#define MSI 3
#define CH  64
#define NRES 5
#define NB  8

// ---------------------------------------------------------------------------
// Implicit-GEMM conv via V_WMMA_F32_16X16X32_F16, tap-major K, NHWC activations.
// X: NHWC f16, channels padded to CPAD (zero-filled).
// Wt: f16 [Cout][KHW][CPAD] (zero-padded), optional per-batch stride.
// 256 threads = 8 wave32; each wave computes a 16-pixel x 32-Cout tile
// (two accumulators sharing each A fragment -> 2 wmma per A load).
// flags: 1 = leaky-relu(0.01), 2 = clamp[0,1]
// ---------------------------------------------------------------------------
template <int KHW_T, int CPAD_T>
__global__ __launch_bounds__(256)
void conv_wmma(const _Float16* __restrict__ Xn, const _Float16* __restrict__ Wt,
               i64 wBatchStride,
               const float* __restrict__ bias,
               const float* __restrict__ bnG, const float* __restrict__ bnB,
               const _Float16* __restrict__ resH,   // NHWC residual (yCpad layout)
               const float*    __restrict__ addF32, // NCHW f32 add (same layout as Yf32)
               float* __restrict__ Yf32,            // NCHW f32 out (optional)
               _Float16* __restrict__ Yc,           // NCHW f16 out (optional)
               _Float16* __restrict__ Yh,           // NHWC f16 out (optional, yCpad)
               int H, int W, int Cout, int stride, int pad,
               int Ho, int Wo, int yCtot, int yCoff, int yCpad, int flags)
{
  constexpr int KW_T   = (KHW_T == 9) ? 3 : (KHW_T == 25 ? 5 : 1);
  constexpr int KPAD_T = KHW_T * CPAD_T;
  constexpr int KVEC_T = KPAD_T / 8;           // 16B chunks per weight row

  const int tid  = threadIdx.x;
  const int lane = tid & 31;
  const int wave = tid >> 5;
  const int bz   = blockIdx.z;
  const int coutBase = blockIdx.y * 32;
  const int pixBase  = blockIdx.x * 128;

  __shared__ alignas(32) _Float16 Bs[32 * KPAD_T];

  // Stage 32 x KPAD weight tile with vectorized 16B copies
  // (row predicate is uniform across each 8-half chunk).
  const _Float16* Wb = Wt + (i64)bz * wBatchStride;
  const v8h zero = {};
  for (int idx = tid; idx < 32 * KVEC_T; idx += 256) {
    int n  = idx / KVEC_T;
    int k8 = idx - n * KVEC_T;
    v8h v = zero;
    if ((coutBase + n) < Cout)
      v = *(const v8h*)(Wb + (i64)(coutBase + n) * KPAD_T + k8 * 8);
    *(v8h*)(&Bs[n * KPAD_T + k8 * 8]) = v;
  }
  __syncthreads();

  const int mBase = pixBase + wave * 16;
  const int myRow = mBase + (lane & 15);
  const int khalf = lane >> 4;
  const bool rowValid = (myRow < Ho * Wo);
  const int oy = rowValid ? (myRow / Wo) : 0;
  const int ox = rowValid ? (myRow - oy * Wo) : 0;
  const _Float16* Xb = Xn + (i64)bz * H * W * CPAD_T;

  __builtin_prefetch(Xb + ((i64)(oy * stride) * W + ox * stride) * CPAD_T, 0, 1);

  const _Float16* bsRow0 = &Bs[(lane & 15) * KPAD_T + khalf * 16];
  const _Float16* bsRow1 = bsRow0 + 16 * KPAD_T;

  v8f acc0 = {}, acc1 = {};
#pragma unroll
  for (int tap = 0; tap < KHW_T; ++tap) {
    const int rr = tap / KW_T, cc = tap - (tap / KW_T) * KW_T;
    const int iy = oy * stride + rr - pad;
    const int ix = ox * stride + cc - pad;
    const bool valid = rowValid && iy >= 0 && iy < H && ix >= 0 && ix < W;
    const _Float16* px = Xb + ((i64)iy * W + ix) * CPAD_T + khalf * 16;
#pragma unroll
    for (int c0 = 0; c0 < CPAD_T; c0 += 32) {
      v8h lo = valid ? *(const v8h*)(px + c0)      : zero;   // k = c0 + khalf*16 + 0..7
      v8h hi = valid ? *(const v8h*)(px + c0 + 8)  : zero;   // k = c0 + khalf*16 + 8..15
      v16h a = __builtin_shufflevector(lo, hi, 0, 1, 2, 3, 4, 5, 6, 7,
                                       8, 9, 10, 11, 12, 13, 14, 15);
      v16h b0 = *(const v16h*)(bsRow0 + tap * CPAD_T + c0);  // contiguous 32B
      v16h b1 = *(const v16h*)(bsRow1 + tap * CPAD_T + c0);
      acc0 = __builtin_amdgcn_wmma_f32_16x16x32_f16(false, a, false, b0,
                                                    (short)0, acc0, false, false);
      acc1 = __builtin_amdgcn_wmma_f32_16x16x32_f16(false, a, false, b1,
                                                    (short)0, acc1, false, false);
    }
  }

  // Epilogue: bias -> BN -> lrelu -> +residual(NHWC) -> +addF32(NCHW) -> clamp -> store
  const int n = lane & 15;
#pragma unroll
  for (int nt = 0; nt < 2; ++nt) {
    const int cout = coutBase + nt * 16 + n;
    float bv = 0.f, sc = 1.f, sh = 0.f;
    if (cout < Cout) {
      if (bias) bv = bias[cout];
      if (bnG)  { sc = bnG[cout] * rsqrtf(1.00001f); sh = bnB[cout]; }
    }
#pragma unroll
    for (int r = 0; r < 8; ++r) {
      int p = mBase + khalf * 8 + r;
      if (p < Ho * Wo) {
        int yy = p / Wo, xx = p - yy * Wo;
        float v = (nt == 0 ? acc0[r] : acc1[r]) + bv;
        v = v * sc + sh;
        if (flags & 1) v = v > 0.f ? v : 0.01f * v;
        i64 oH = (((i64)bz * Ho + yy) * (i64)Wo + xx) * yCpad + cout;
        if (resH) v += (float)resH[oH];
        if (cout < Cout) {
          i64 oN = (((i64)bz * yCtot + yCoff + cout) * Ho + yy) * (i64)Wo + xx;
          float vn = v;
          if (addF32) vn += addF32[oN];
          if (flags & 2) vn = fminf(fmaxf(vn, 0.f), 1.f);
          if (Yf32) Yf32[oN] = vn;
          if (Yc)   Yc[oN]   = (_Float16)vn;
        }
        if (Yh && cout < yCpad) Yh[oH] = (_Float16)v;  // padded couts give v==0
      }
    }
  }
}

// ---------------------------------------------------------------------------
// Helper kernels
// ---------------------------------------------------------------------------
// NCHW f32 -> NHWC f16 (channel-padded, zero fill), with shift_window offset.
// r0=c0=1 gives the identity copy; out[y][x] = in[y+r0-1][x+c0-1] else 0.
__global__ void to_nhwc(const float* __restrict__ in, _Float16* __restrict__ out,
                        int C, int Cpad, int H, int W, int r0, int c0) {
  i64 i = (i64)blockIdx.x * 256 + threadIdx.x;
  i64 n = (i64)NB * H * W * Cpad;
  if (i >= n) return;
  int c = i % Cpad; i64 t = i / Cpad; int x = t % W; t /= W; int y = t % H; int b = (int)(t / H);
  float v = 0.f;
  if (c < C) {
    int sy = y + r0 - 1, sx = x + c0 - 1;
    if (sy >= 0 && sy < H && sx >= 0 && sx < W)
      v = in[(((i64)b * C + c) * H + sy) * W + sx];
  }
  out[i] = (_Float16)v;
}

// f32 [O][I][KH][KW] -> f16 [O][KHW][Cpad] (zero-padded channels)
__global__ void prep_w(const float* __restrict__ in, _Float16* __restrict__ out,
                       int O, int I, int KHW, int Cpad) {
  i64 i = (i64)blockIdx.x * 256 + threadIdx.x;
  i64 n = (i64)O * KHW * Cpad;
  if (i >= n) return;
  int c = i % Cpad; i64 t = i / Cpad; int tap = t % KHW; int o = (int)(t / KHW);
  float v = 0.f;
  if (c < I) v = in[((i64)o * I + c) * KHW + tap];
  out[i] = (_Float16)v;
}

__global__ void fill0_f16(_Float16* __restrict__ p, i64 n) {
  i64 i = (i64)blockIdx.x * 256 + threadIdx.x;
  if (i < n) p[i] = (_Float16)0.f;
}

// correlation kernels Kq: out[b][p][tap][64] from hsi_f NHWC(64)
__global__ void make_corr_w(const _Float16* __restrict__ fN, _Float16* __restrict__ out) {
  i64 i = (i64)blockIdx.x * 256 + threadIdx.x;
  i64 n = (i64)NB * 25 * 9 * 64;
  if (i >= n) return;
  int c = i % 64; i64 t = i / 64; int tap = t % 9; t /= 9; int p = t % 25; int b = (int)(t / 25);
  int py = p / 5, px = p % 5, dy = tap / 3, dx = tap % 3;
  out[i] = fN[(((i64)b * 64 + py * 3 + dy) * 64 + (px * 3 + dx)) * 64 + c];
}

// transposed-conv kernels: out[b][o(31)][tap][32] = hsi_low NHWC(32) flipped/swapped
__global__ void make_tconv_w(const _Float16* __restrict__ lowN, _Float16* __restrict__ out) {
  i64 i = (i64)blockIdx.x * 256 + threadIdx.x;
  i64 n = (i64)NB * HSI * 9 * 32;
  if (i >= n) return;
  int p = i % 32; i64 t = i / 32; int tap = t % 9; t /= 9; int o = t % HSI; int b = (int)(t / HSI);
  _Float16 v = (_Float16)0.f;
  if (p < 25) {
    int py = p / 5, px = p % 5, dy = 2 - tap / 3, dx = 2 - tap % 3;
    v = lowN[(((i64)b * 64 + py * 3 + dy) * 64 + (px * 3 + dx)) * 32 + o];
  }
  out[i] = v;
}

// softmax over last dim of NCHW f32 (in place) + emit NHWC f16 (Cpad)
__global__ void softmax_lastdim(float* __restrict__ data, _Float16* __restrict__ outH,
                                i64 rows, int C, int HH, int L, int Cpad) {
  i64 row = (i64)blockIdx.x * (blockDim.x / 32) + (threadIdx.x >> 5);
  int lane = threadIdx.x & 31;
  if (row >= rows) return;
  float* p = data + row * L;
  int y = (int)(row % HH); i64 t = row / HH; int c = (int)(t % C); int b = (int)(t / C);
  float mx = -1e30f;
  for (int i = lane; i < L; i += 32) mx = fmaxf(mx, p[i]);
  for (int o = 16; o > 0; o >>= 1) mx = fmaxf(mx, __shfl_xor(mx, o, 32));
  float s = 0.f;
  for (int i = lane; i < L; i += 32) s += __expf(p[i] - mx);
  for (int o = 16; o > 0; o >>= 1) s += __shfl_xor(s, o, 32);
  float inv = 1.f / s;
  for (int i = lane; i < L; i += 32) {
    float v = __expf(p[i] - mx) * inv;
    p[i] = v;
    outH[(((i64)b * HH + y) * L + i) * Cpad + c] = (_Float16)v;
  }
}

// depthwise 11x11 pad 5: NCHW f16 in -> NHWC f16 out (padded to Cpad)
__global__ void dwconv11(const _Float16* __restrict__ X, const float* __restrict__ Wd,
                         const float* __restrict__ bias, _Float16* __restrict__ Yn,
                         int C, int Cpad, int H, int W) {
  i64 i = (i64)blockIdx.x * 256 + threadIdx.x;
  i64 n = (i64)NB * H * W * Cpad;
  if (i >= n) return;
  int c = i % Cpad; i64 t = i / Cpad; int x = t % W; t /= W; int y = t % H; int b = (int)(t / H);
  float acc = 0.f;
  if (c < C) {
    const float* w = Wd + (i64)c * 121;
    const _Float16* xb = X + (((i64)b * C + c) * H) * W;
    acc = bias[c];
    for (int r = 0; r < 11; ++r) {
      int iy = y + r - 5; if (iy < 0 || iy >= H) continue;
      for (int s = 0; s < 11; ++s) {
        int ix = x + s - 5; if (ix < 0 || ix >= W) continue;
        acc += (float)xb[iy * W + ix] * w[r * 11 + s];
      }
    }
  }
  Yn[i] = (_Float16)acc;
}

__device__ inline float cubw(float x) {           // Catmull-Rom (a = -0.5)
  x = fabsf(x);
  if (x < 1.f) return 1.5f * x * x * x - 2.5f * x * x + 1.f;
  if (x < 2.f) return -0.5f * x * x * x + 2.5f * x * x - 4.f * x + 2.f;
  return 0.f;
}

__global__ void bicubic4(const float* __restrict__ in, float* __restrict__ out,
                         int C, int Hi, int Wi) {
  int Ho = Hi * 4, Wo = Wi * 4;
  i64 i = (i64)blockIdx.x * 256 + threadIdx.x;
  i64 n = (i64)NB * C * Ho * Wo;
  if (i >= n) return;
  int x = i % Wo; i64 t = i / Wo; int y = t % Ho; i64 bc = t / Ho;
  float sy = (y + 0.5f) * 0.25f - 0.5f;
  float sx = (x + 0.5f) * 0.25f - 0.5f;
  int y0 = (int)floorf(sy) - 1, x0 = (int)floorf(sx) - 1;
  const float* src = in + bc * Hi * Wi;
  float acc = 0.f;
  for (int r = 0; r < 4; ++r) {
    int yy = min(max(y0 + r, 0), Hi - 1);
    float wy = cubw(sy - (float)(y0 + r));
    for (int s = 0; s < 4; ++s) {
      int xx = min(max(x0 + s, 0), Wi - 1);
      acc += wy * cubw(sx - (float)(x0 + s)) * src[yy * Wi + xx];
    }
  }
  out[i] = acc;
}

// simi[b,c,k] = softmax_k( sum_n mf[b,c,n] * hf[b,k,n] )
__global__ void spectral_attn(const float* __restrict__ mf, const float* __restrict__ hf,
                              float* __restrict__ simi) {
  int b = blockIdx.x, c = blockIdx.y, lane = threadIdx.x;
  __shared__ float row[31];
  for (int k = lane; k < 31; k += 32) {
    const float* m = mf + ((i64)b * 3 + c) * 256;
    const float* h = hf + ((i64)b * 31 + k) * 256;
    float s = 0.f;
    for (int nn = 0; nn < 256; ++nn) s += m[nn] * h[nn];
    row[k] = s;
  }
  __syncthreads();
  float v = (lane < 31) ? row[lane] : -1e30f;
  float mx = v;
  for (int o = 16; o > 0; o >>= 1) mx = fmaxf(mx, __shfl_xor(mx, o, 32));
  float e = (lane < 31) ? __expf(v - mx) : 0.f;
  float s = e;
  for (int o = 16; o > 0; o >>= 1) s += __shfl_xor(s, o, 32);
  if (lane < 31) simi[((i64)b * 3 + c) * 31 + lane] = e / s;
}

// feat[b, Coff+k, n] = sum_c simi[b,c,k] * align[b,c,n]  (feat NCHW f16)
__global__ void spectral_apply(const float* __restrict__ simi, const _Float16* __restrict__ align,
                               _Float16* __restrict__ feat, int HW, int Ctot, int Coff) {
  i64 i = (i64)blockIdx.x * 256 + threadIdx.x;
  i64 n = (i64)NB * 31 * HW;
  if (i >= n) return;
  int pos = i % HW; i64 t = i / HW; int k = t % 31; int b = (int)(t / 31);
  float s = 0.f;
  for (int c = 0; c < 3; ++c)
    s += simi[((i64)b * 3 + c) * 31 + k] * (float)align[((i64)b * 3 + c) * HW + pos];
  feat[(((i64)b * Ctot + Coff + k) * (i64)HW) + pos] = (_Float16)s;
}

// ---------------------------------------------------------------------------
// Host orchestration
// ---------------------------------------------------------------------------
static void launch_conv(hipStream_t s, int KHW, int CPAD,
                        const _Float16* X, const _Float16* Wt, i64 wbs,
                        const float* bias, const float* bng, const float* bnb,
                        const _Float16* resH, const float* add32,
                        float* y32, _Float16* yc, _Float16* yh,
                        int H, int W, int Cout, int st, int pad,
                        int yCtot, int yCoff, int yCpad, int flags) {
  int KH = (KHW == 9) ? 3 : (KHW == 25 ? 5 : 1);
  int Ho = (H + 2 * pad - KH) / st + 1;
  int Wo = (W + 2 * pad - KH) / st + 1;
  int Cgrid = yh ? ((yCpad + 31) & ~31) : ((Cout + 31) & ~31);
  dim3 g((Ho * Wo + 127) / 128, Cgrid / 32, NB);
  if (KHW == 9 && CPAD == 32)
    conv_wmma<9, 32><<<g, 256, 0, s>>>(X, Wt, wbs, bias, bng, bnb, resH, add32, y32, yc, yh,
                                       H, W, Cout, st, pad, Ho, Wo, yCtot, yCoff, yCpad, flags);
  else if (KHW == 9 && CPAD == 64)
    conv_wmma<9, 64><<<g, 256, 0, s>>>(X, Wt, wbs, bias, bng, bnb, resH, add32, y32, yc, yh,
                                       H, W, Cout, st, pad, Ho, Wo, yCtot, yCoff, yCpad, flags);
  else if (KHW == 25 && CPAD == 32)
    conv_wmma<25, 32><<<g, 256, 0, s>>>(X, Wt, wbs, bias, bng, bnb, resH, add32, y32, yc, yh,
                                        H, W, Cout, st, pad, Ho, Wo, yCtot, yCoff, yCpad, flags);
  else
    conv_wmma<1, 160><<<g, 256, 0, s>>>(X, Wt, wbs, bias, bng, bnb, resH, add32, y32, yc, yh,
                                        H, W, Cout, st, pad, Ho, Wo, yCtot, yCoff, yCpad, flags);
}

extern "C" void kernel_launch(void* const* d_in, const int* in_sizes, int n_in,
                              void* d_out, int out_size, void* d_ws, size_t ws_size,
                              hipStream_t stream) {
  (void)in_sizes; (void)n_in; (void)out_size; (void)ws_size;
  const float* LRHSI = (const float*)d_in[0];
  const float* HRMSI = (const float*)d_in[1];
  const float* agg_dw_b = (const float*)d_in[2];
  const float* agg_dw_w = (const float*)d_in[3];
  const float* agg_pw_b = (const float*)d_in[4];
  const float* agg_pw_w = (const float*)d_in[5];
  const float* h_bn1b = (const float*)d_in[6];
  const float* h_bn1g = (const float*)d_in[7];
  const float* h_bn2b = (const float*)d_in[8];
  const float* h_bn2g = (const float*)d_in[9];
  const float* h_c1b  = (const float*)d_in[10];
  const float* h_c1w  = (const float*)d_in[11];
  const float* h_c2b  = (const float*)d_in[12];
  const float* h_c2w  = (const float*)d_in[13];
  const float* h_rw1  = (const float*)d_in[14];
  const float* h_rw2  = (const float*)d_in[15];
  const float* low_b  = (const float*)d_in[16];
  const float* low_w  = (const float*)d_in[17];
  const float* m_bn1b = (const float*)d_in[18];
  const float* m_bn1g = (const float*)d_in[19];
  const float* m_bn2b = (const float*)d_in[20];
  const float* m_bn2g = (const float*)d_in[21];
  const float* m_c1b  = (const float*)d_in[22];
  const float* m_c1w  = (const float*)d_in[23];
  const float* m_c2b  = (const float*)d_in[24];
  const float* m_c2w  = (const float*)d_in[25];
  const float* m_rw1  = (const float*)d_in[26];
  const float* m_rw2  = (const float*)d_in[27];
  const float* align_b = (const float*)d_in[28];
  const float* align_w = (const float*)d_in[29];
  const float* hsid_b  = (const float*)d_in[30];
  const float* hsid_w  = (const float*)d_in[31];
  const float* msid_b  = (const float*)d_in[32];
  const float* msid_w  = (const float*)d_in[33];

  char* ws = (char*)d_ws; size_t off = 0;
  auto alloc = [&](size_t bytes) -> void* {
    void* p = ws + off; off = (off + bytes + 255) & ~(size_t)255; return p;
  };
  auto prepw = [&](const float* src, _Float16* dst, int O, int I, int KHW, int CPAD) {
    i64 n = (i64)O * KHW * CPAD;
    prep_w<<<(int)((n + 255) / 256), 256, 0, stream>>>(src, dst, O, I, KHW, CPAD);
  };

  const int HW64 = 64 * 64, HW256 = 256 * 256;
  // NHWC activations (channel-padded)
  _Float16* lrN      = (_Float16*)alloc((i64)NB * HW64 * 32 * 2);
  _Float16* hrN      = (_Float16*)alloc((i64)NB * HW256 * 32 * 2);     // 33.5 MB
  _Float16* tmpA     = (_Float16*)alloc((i64)NB * HW256 * 64 * 2);     // 67 MB
  _Float16* msiF     = (_Float16*)alloc((i64)NB * HW256 * 64 * 2);     // 67 MB (persistent)
  _Float16* hsiShiftN= (_Float16*)alloc((i64)NB * HW64 * 32 * 2);
  _Float16* hsiT0    = (_Float16*)alloc((i64)NB * HW64 * 64 * 2);
  _Float16* hsiT1    = (_Float16*)alloc((i64)NB * HW64 * 64 * 2);
  _Float16* hsiLowN  = (_Float16*)alloc((i64)NB * HW64 * 32 * 2);
  _Float16* corrW2   = (_Float16*)alloc((i64)NB * 25 * 9 * 64 * 2);
  _Float16* tconvW2  = (_Float16*)alloc((i64)NB * HSI * 9 * 32 * 2);
  float*    simiF32  = (float*)   alloc((i64)NB * 25 * 254 * 254 * 4); // 51.6 MB
  _Float16* simiH    = (_Float16*)alloc((i64)NB * 254 * 254 * 32 * 2); // 33 MB NHWC
  _Float16* featF16  = (_Float16*)alloc((i64)NB * 5 * HSI * HW256 * 2);// 162 MB NCHW
  _Float16* alignF16 = (_Float16*)alloc((i64)NB * MSI * HW256 * 2);    // NCHW
  float*    hfF32    = (float*)   alloc((i64)NB * HSI * 256 * 4);
  float*    mfF32    = (float*)   alloc((i64)NB * MSI * 256 * 4);
  float*    attn     = (float*)   alloc((i64)NB * 3 * 31 * 4);
  float*    upF32    = (float*)   alloc((i64)NB * HSI * HW256 * 4);    // 65 MB
  _Float16* dwN      = (_Float16*)alloc((i64)NB * HW256 * 160 * 2);    // 168 MB NHWC
  // f16 weights, layout [O][KHW][CPAD]
  _Float16* w_hc1  = (_Float16*)alloc((i64)HSI * 9 * 32 * 2);
  _Float16* w_hc2  = (_Float16*)alloc((i64)CH * 9 * 32 * 2);
  _Float16* w_hr1  = (_Float16*)alloc((i64)NRES * CH * 9 * 64 * 2);
  _Float16* w_hr2  = (_Float16*)alloc((i64)NRES * CH * 9 * 64 * 2);
  _Float16* w_mc1  = (_Float16*)alloc((i64)MSI * 9 * 32 * 2);
  _Float16* w_mc2  = (_Float16*)alloc((i64)CH * 9 * 32 * 2);
  _Float16* w_mr1  = (_Float16*)alloc((i64)NRES * CH * 9 * 64 * 2);
  _Float16* w_mr2  = (_Float16*)alloc((i64)NRES * CH * 9 * 64 * 2);
  _Float16* w_low  = (_Float16*)alloc((i64)HSI * 9 * 32 * 2);
  _Float16* w_algn = (_Float16*)alloc((i64)MSI * 9 * 32 * 2);
  _Float16* w_hsid = (_Float16*)alloc((i64)HSI * 25 * 32 * 2);
  _Float16* w_msid = (_Float16*)alloc((i64)MSI * 25 * 32 * 2);
  _Float16* w_pw   = (_Float16*)alloc((i64)HSI * 1 * 160 * 2);

  // -- weight prep (OIHW f32 -> [O][tap][Cpad] f16)
  prepw(h_c1w, w_hc1, HSI, HSI, 9, 32);
  prepw(h_c2w, w_hc2, CH, HSI, 9, 32);
  prepw(h_rw1, w_hr1, NRES * CH, CH, 9, 64);   // stacked: 5*64 output rows
  prepw(h_rw2, w_hr2, NRES * CH, CH, 9, 64);
  prepw(m_c1w, w_mc1, MSI, MSI, 9, 32);
  prepw(m_c2w, w_mc2, CH, MSI, 9, 32);
  prepw(m_rw1, w_mr1, NRES * CH, CH, 9, 64);
  prepw(m_rw2, w_mr2, NRES * CH, CH, 9, 64);
  prepw(low_w, w_low, HSI, HSI, 9, 32);
  prepw(align_w, w_algn, MSI, MSI, 9, 32);
  prepw(hsid_w, w_hsid, HSI, HSI, 25, 32);
  prepw(msid_w, w_msid, MSI, MSI, 25, 32);
  prepw(agg_pw_w, w_pw, HSI, 5 * HSI, 1, 160);

  // -- NHWC input staging
  { i64 n = (i64)NB * HW64 * 32;
    to_nhwc<<<(int)((n + 255) / 256), 256, 0, stream>>>(LRHSI, lrN, HSI, 32, 64, 64, 1, 1); }
  { i64 n = (i64)NB * HW256 * 32;
    to_nhwc<<<(int)((n + 255) / 256), 256, 0, stream>>>(HRMSI, hrN, MSI, 32, 256, 256, 1, 1); }

  // -- cubic x4 upsample of LRHSI (residual base)
  { i64 n = (i64)NB * HSI * HW256;
    bicubic4<<<(int)((n + 255) / 256), 256, 0, stream>>>(LRHSI, upF32, HSI, 64, 64); }

  // -- MSI encoder: computed ONCE (HRMSI is not shifted) -> msiF (NHWC,64)
  launch_conv(stream, 9, 32, hrN, w_mc1, 0, m_c1b, m_bn1g, m_bn1b, nullptr, nullptr,
              nullptr, nullptr, tmpA, 256, 256, MSI, 1, 1, 0, 0, 32, 1);
  launch_conv(stream, 9, 32, tmpA, w_mc2, 0, m_c2b, m_bn2g, m_bn2b, nullptr, nullptr,
              nullptr, nullptr, msiF, 256, 256, CH, 1, 1, 0, 0, 64, 0);
  for (int i = 0; i < NRES; ++i) {
    launch_conv(stream, 9, 64, msiF, w_mr1 + (i64)i * CH * 9 * 64, 0, nullptr, nullptr, nullptr,
                nullptr, nullptr, nullptr, nullptr, tmpA, 256, 256, CH, 1, 1, 0, 0, 64, 1);
    launch_conv(stream, 9, 64, tmpA, w_mr2 + (i64)i * CH * 9 * 64, 0, nullptr, nullptr, nullptr,
                msiF, nullptr, nullptr, nullptr, msiF, 256, 256, CH, 1, 1, 0, 0, 64, 0);
  }

  // -- spectral block
  launch_conv(stream, 9, 32, hrN, w_algn, 0, align_b, nullptr, nullptr, nullptr, nullptr,
              nullptr, alignF16, nullptr, 256, 256, MSI, 1, 1, MSI, 0, 0, 1);
  launch_conv(stream, 25, 32, lrN, w_hsid, 0, hsid_b, nullptr, nullptr, nullptr, nullptr,
              hfF32, nullptr, nullptr, 64, 64, HSI, 4, 1, HSI, 0, 0, 1);
  launch_conv(stream, 25, 32, hrN, w_msid, 0, msid_b, nullptr, nullptr, nullptr, nullptr,
              mfF32, nullptr, nullptr, 256, 256, MSI, 16, 0, MSI, 0, 0, 1);
  spectral_attn<<<dim3(NB, 3), 32, 0, stream>>>(mfF32, hfF32, attn);
  { i64 n = (i64)NB * 31 * HW256;
    spectral_apply<<<(int)((n + 255) / 256), 256, 0, stream>>>(attn, alignF16, featF16,
                                                               HW256, 5 * HSI, 4 * HSI); }

  // -- 4 shifted spatial blocks
  { i64 n = (i64)NB * 254 * 254 * 32;      // zero padded channels of simiH once per call
    fill0_f16<<<(int)((n + 255) / 256), 256, 0, stream>>>(simiH, n); }
  const int R0[4] = {1, 0, 1, 0}, C0[4] = {1, 1, 0, 0};
  for (int t = 0; t < 4; ++t) {
    { i64 n = (i64)NB * HW64 * 32;
      to_nhwc<<<(int)((n + 255) / 256), 256, 0, stream>>>(LRHSI, hsiShiftN, HSI, 32, 64, 64,
                                                          R0[t], C0[t]); }
    // HSI encoder
    launch_conv(stream, 9, 32, hsiShiftN, w_hc1, 0, h_c1b, h_bn1g, h_bn1b, nullptr, nullptr,
                nullptr, nullptr, hsiT0, 64, 64, HSI, 1, 1, 0, 0, 32, 1);
    launch_conv(stream, 9, 32, hsiT0, w_hc2, 0, h_c2b, h_bn2g, h_bn2b, nullptr, nullptr,
                nullptr, nullptr, hsiT1, 64, 64, CH, 1, 1, 0, 0, 64, 0);
    for (int i = 0; i < NRES; ++i) {
      launch_conv(stream, 9, 64, hsiT1, w_hr1 + (i64)i * CH * 9 * 64, 0, nullptr, nullptr,
                  nullptr, nullptr, nullptr, nullptr, nullptr, hsiT0, 64, 64, CH, 1, 1,
                  0, 0, 64, 1);
      launch_conv(stream, 9, 64, hsiT0, w_hr2 + (i64)i * CH * 9 * 64, 0, nullptr, nullptr,
                  nullptr, hsiT1, nullptr, nullptr, nullptr, hsiT1, 64, 64, CH, 1, 1,
                  0, 0, 64, 0);
    }
    // hsi_low
    launch_conv(stream, 9, 32, hsiShiftN, w_low, 0, low_b, nullptr, nullptr, nullptr, nullptr,
                nullptr, nullptr, hsiLowN, 64, 64, HSI, 1, 1, 0, 0, 32, 1);
    // patch kernels
    { i64 n = (i64)NB * 25 * 9 * 64;
      make_corr_w<<<(int)((n + 255) / 256), 256, 0, stream>>>(hsiT1, corrW2); }
    { i64 n = (i64)NB * HSI * 9 * 32;
      make_tconv_w<<<(int)((n + 255) / 256), 256, 0, stream>>>(hsiLowN, tconvW2); }
    // correlation conv: msi_f (x) per-batch kernels, valid -> 254x254 (NCHW f32)
    launch_conv(stream, 9, 64, msiF, corrW2, (i64)25 * 9 * 64, nullptr, nullptr, nullptr,
                nullptr, nullptr, simiF32, nullptr, nullptr, 256, 256, 25, 1, 0, 25, 0, 0, 0);
    // softmax over width -> NHWC f16
    { i64 rows = (i64)NB * 25 * 254;
      softmax_lastdim<<<(int)((rows + 7) / 8), 256, 0, stream>>>(simiF32, simiH, rows,
                                                                 25, 254, 254, 32); }
    // transposed conv, pad 2 -> 256x256, into feat slab t (NCHW f16)
    launch_conv(stream, 9, 32, simiH, tconvW2, (i64)HSI * 9 * 32, nullptr, nullptr, nullptr,
                nullptr, nullptr, nullptr, featF16, nullptr, 254, 254, HSI, 1, 2,
                5 * HSI, t * HSI, 0, 0);
  }

  // -- aggregation: depthwise 11x11 (NCHW->NHWC160), then pointwise + up + clamp
  { i64 n = (i64)NB * HW256 * 160;
    dwconv11<<<(int)((n + 255) / 256), 256, 0, stream>>>(featF16, agg_dw_w, agg_dw_b, dwN,
                                                         5 * HSI, 160, 256, 256); }
  launch_conv(stream, 1, 160, dwN, w_pw, 0, agg_pw_b, nullptr, nullptr, nullptr, upF32,
              (float*)d_out, nullptr, nullptr, 256, 256, HSI, 1, 0, HSI, 0, 0, 2);
}
